// Attention_27814208209106
// MI455X (gfx1250) — matrix operand order repaired and checked by reading, not compiled, not run on hardware
//
#include <hip/hip_runtime.h>
#include <math.h>

// ---------------- problem constants ----------------
#define B_    2
#define T_    1024
#define S_    2048
#define EMB_  3584
#define NH_   28
#define KVH_  4
#define HD_   128
#define NREP_ 7
#define QKD_  (NH_*HD_)   // 3584
#define KVD_  (KVH_*HD_)  // 512
#define M_    (B_*T_)     // 2048
#define QB_   (T_/64)     // 16 q-blocks of 64 rows
#define KMASK_ (-3.3895313892515355e38f)  // bf16 min, as in reference

typedef _Float16 half8 __attribute__((ext_vector_type(8)));
typedef _Float16 v16h  __attribute__((ext_vector_type(16)));
typedef float    v8f   __attribute__((ext_vector_type(8)));

__device__ __forceinline__ v16h hcat(half8 a, half8 b) {
  return __builtin_shufflevector(a, b, 0,1,2,3,4,5,6,7,8,9,10,11,12,13,14,15);
}
__device__ __forceinline__ v8f wmma16(v16h a, v16h b, v8f c) {
  // D = A(16x32 f16) * B(32x16 f16) + C(16x16 f32)
  return __builtin_amdgcn_wmma_f32_16x16x32_f16(false, a, false, b, (short)0, c, false, false);
}

// ---- CDNA5 async global->LDS (ASYNCcnt path, ISA 15.18.3 op 98) ----
// VDST = VGPR with LDS byte address (low 32 bits of generic LDS pointer),
// VADDR = 64-bit global address, saddr=off. 16 bytes per lane.
__device__ __forceinline__ void async_g2l_b128(unsigned lds_off, const void* gptr) {
  asm volatile("global_load_async_to_lds_b128 %0, %1, off"
               :: "v"(lds_off), "v"(gptr)
               : "memory");
}
__device__ __forceinline__ void wait_async0() {
  asm volatile("s_wait_asynccnt 0x0" ::: "memory");
}
__device__ __forceinline__ unsigned lds_off32(const void* p) {
  return (unsigned)(unsigned long long)p;  // LDS byte address = low 32 bits of the pointer
}

// ---------------- elementwise converters ----------------
__global__ __launch_bounds__(256) void cvt_f16(const float* __restrict__ src,
                                               _Float16* __restrict__ dst, size_t n) {
  size_t i = (size_t)blockIdx.x * 256 + threadIdx.x;
  if (i < n) dst[i] = (_Float16)src[i];
}

// src: (K,N) row-major f32 -> dst: (N,K) row-major f16
__global__ __launch_bounds__(256) void cvt_transpose(const float* __restrict__ src,
                                                     _Float16* __restrict__ dst,
                                                     int K, int N) {
  size_t i = (size_t)blockIdx.x * 256 + threadIdx.x;
  if (i < (size_t)K * N) {
    int k = (int)(i / N), n = (int)(i % N);
    dst[(size_t)n * K + k] = (_Float16)src[i];
  }
}

// ---------------- positions / left pads ----------------
__global__ void compute_pos(const int* __restrict__ seg, const int* __restrict__ curp,
                            int* __restrict__ pos, int* __restrict__ startind) {
  int b = blockIdx.x;
  if (threadIdx.x != 0) return;
  int cur = *curp;
  int csum = 0, lp = 0;
  for (int t = 0; t < T_; ++t) {
    int v = (seg[b * T_ + t] != 0) ? 1 : 0;
    csum += v;
    if (csum == 0) lp++;
    pos[b * T_ + t] = csum - 1 + cur;
  }
  startind[b] = lp;
}

// ---------------- tiled WMMA GEMM: C(MxN,f32) = A(MxK,f16) * Bt(NxK,f16)^T + bias ----
// Double-buffered: tiles stream into LDS via global_load_async_to_lds_b128 while the
// previous tile is consumed by WMMA; one barrier per k-step.
#define BK_ 32
#define LDT 40   // padded LDS row stride (halves): conflict-free frag reads

__global__ __launch_bounds__(256) void gemm_f16f32(const _Float16* __restrict__ A,
                                                   const _Float16* __restrict__ Bt,
                                                   const float* __restrict__ bias,
                                                   float* __restrict__ C,
                                                   int M, int N, int K) {
  __shared__ __align__(16) _Float16 As[2][128 * LDT];
  __shared__ __align__(16) _Float16 Bs[2][128 * LDT];
  const int tid = threadIdx.x, lane = tid & 31, w = tid >> 5;
  const int wr = w >> 1, wc = w & 1;            // 4x2 wave grid -> 32x64 per wave
  const int hl = lane >> 4, l15 = lane & 15;
  const int m0 = blockIdx.y * 128, n0 = blockIdx.x * 128;
  const int lr = tid >> 1, lc = (tid & 1) * 16; // 256 thr load 128x32 halves

  v8f acc[2][4];
  for (int i = 0; i < 2; ++i)
    for (int j = 0; j < 4; ++j)
      for (int r = 0; r < 8; ++r) acc[i][j][r] = 0.0f;

  const _Float16* ag = A  + (size_t)(m0 + lr) * K + lc;
  const _Float16* bg = Bt + (size_t)(n0 + lr) * K + lc;
  unsigned aoff[2], boff[2];
  for (int u = 0; u < 2; ++u) {
    aoff[u] = lds_off32(&As[u][lr * LDT + lc]);
    boff[u] = lds_off32(&Bs[u][lr * LDT + lc]);
  }

  // prologue: stream first tile
  async_g2l_b128(aoff[0],      ag);
  async_g2l_b128(aoff[0] + 16, ag + 8);
  async_g2l_b128(boff[0],      bg);
  async_g2l_b128(boff[0] + 16, bg + 8);

  const int nk = K / BK_;
  for (int it = 0; it < nk; ++it) {
    wait_async0();        // my slice of buf[it&1] landed in LDS
    __syncthreads();      // everyone's slice landed; prior reads of buf[(it+1)&1] done
    const int cb = it & 1;
    if (it + 1 < nk) {    // overlap next tile's DMA with this tile's WMMAs
      const int nb = cb ^ 1;
      const int kn = (it + 1) * BK_;
      async_g2l_b128(aoff[nb],      ag + kn);
      async_g2l_b128(aoff[nb] + 16, ag + kn + 8);
      async_g2l_b128(boff[nb],      bg + kn);
      async_g2l_b128(boff[nb] + 16, bg + kn + 8);
      if (it + 2 < nk) {  // far prefetch into GL2
        __builtin_prefetch(ag + (it + 2) * BK_, 0, 1);
        __builtin_prefetch(bg + (it + 2) * BK_, 0, 1);
      }
    }

    v16h af[2], bf[4];
    for (int i = 0; i < 2; ++i) {
      // A 16x32 frag: lane<16 holds K 0-7 & 16-23; lane>=16 holds 8-15 & 24-31
      const _Float16* p = &As[cb][(wr * 32 + i * 16 + l15) * LDT];
      af[i] = hcat(*(const half8*)(p + hl * 8), *(const half8*)(p + 16 + hl * 8));
    }
    for (int j = 0; j < 4; ++j) {
      // B 32x16 frag: lane<16 holds K 0-7 & 16-23; lane>=16 holds 8-15 & 24-31
      const _Float16* p = &Bs[cb][(wc * 64 + j * 16 + l15) * LDT];
      bf[j] = hcat(*(const half8*)(p + hl * 8), *(const half8*)(p + 16 + hl * 8));
    }
    for (int i = 0; i < 2; ++i)
      for (int j = 0; j < 4; ++j)
        acc[i][j] = wmma16(af[i], bf[j], acc[i][j]);
  }

  for (int i = 0; i < 2; ++i)
    for (int j = 0; j < 4; ++j) {
      int cc = n0 + wc * 64 + j * 16 + l15;
      float bb = bias ? bias[cc] : 0.0f;
      for (int r = 0; r < 8; ++r) {
        int rr = m0 + wr * 32 + i * 16 + r + hl * 8; // C frag: VGPR r = row r+8*hl
        C[(size_t)rr * N + cc] = acc[i][j][r] + bb;
      }
    }
}

// ---------------- RoPE + KV scatter ----------------
__global__ __launch_bounds__(256) void rope_write(const float* __restrict__ qraw,
                                                  const float* __restrict__ kraw,
                                                  const float* __restrict__ vraw,
                                                  const int* __restrict__ pos,
                                                  const int* __restrict__ curp,
                                                  _Float16* __restrict__ Qh,
                                                  _Float16* __restrict__ Kh,
                                                  _Float16* __restrict__ Vh) {
  int bt = blockIdx.x;
  int b = bt / T_, t = bt % T_;
  float p = (float)pos[bt];
  int cur = *curp;
  int ds = cur + t;
  for (int idx = threadIdx.x; idx < NH_ * 64; idx += 256) {
    int n = idx >> 6, d = idx & 63;
    float ang = p * __powf(1000000.0f, -(float)d * (1.0f / 64.0f));
    float sn, cs; __sincosf(ang, &sn, &cs);
    size_t base = ((size_t)bt * NH_ + n) * HD_;
    float x1 = qraw[base + d], x2 = qraw[base + 64 + d];
    Qh[base + d]      = (_Float16)(x1 * cs - x2 * sn);
    Qh[base + 64 + d] = (_Float16)(x2 * cs + x1 * sn);
  }
  if (ds >= S_) return;
  for (int idx = threadIdx.x; idx < KVH_ * 64; idx += 256) {
    int hh = idx >> 6, d = idx & 63;
    float ang = p * __powf(1000000.0f, -(float)d * (1.0f / 64.0f));
    float sn, cs; __sincosf(ang, &sn, &cs);
    size_t sb = ((size_t)bt * KVH_ + hh) * HD_;
    size_t db = ((size_t)(b * S_ + ds) * KVH_ + hh) * HD_;
    float x1 = kraw[sb + d], x2 = kraw[sb + 64 + d];
    Kh[db + d]      = (_Float16)(x1 * cs - x2 * sn);
    Kh[db + 64 + d] = (_Float16)(x2 * cs + x1 * sn);
  }
  for (int idx = threadIdx.x; idx < KVH_ * HD_; idx += 256) {
    int hh = idx >> 7, d = idx & 127;
    Vh[((size_t)(b * S_ + ds) * KVH_ + hh) * HD_ + d] =
        (_Float16)vraw[((size_t)bt * KVH_ + hh) * HD_ + d];
  }
}

// import untouched cache rows (do not mutate the input caches)
__global__ __launch_bounds__(256) void fill_cache(const float* __restrict__ kc,
                                                  const float* __restrict__ vc,
                                                  const int* __restrict__ curp,
                                                  _Float16* __restrict__ Kh,
                                                  _Float16* __restrict__ Vh) {
  size_t idx = (size_t)blockIdx.x * 256 + threadIdx.x;
  const size_t total = (size_t)B_ * S_ * KVH_ * HD_;
  if (idx >= total) return;
  int cur = *curp;
  int s = (int)((idx / (KVH_ * HD_)) % S_);
  if (s < cur || s >= cur + T_) {
    Kh[idx] = (_Float16)kc[idx];
    Vh[idx] = (_Float16)vc[idx];
  }
}

// ---------------- flash attention (GQA, causal + segment mask) ----------------
#define KLD 136
#define VLD 40
#define PLD 40

__global__ __launch_bounds__(128) void flash_attn(const _Float16* __restrict__ Qh,
                                                  const _Float16* __restrict__ Kh,
                                                  const _Float16* __restrict__ Vh,
                                                  const int* __restrict__ seg,
                                                  const int* __restrict__ startind,
                                                  const int* __restrict__ curp,
                                                  _Float16* __restrict__ Oh) {
  __shared__ __align__(16) _Float16 Ks[32 * KLD];       // K tile: 32 s-rows x 128 d
  __shared__ __align__(16) _Float16 Vt[128 * VLD];      // V tile transposed: 128 d x 32 s
  __shared__ __align__(16) _Float16 Pt[4 * 16 * PLD];   // per-wave P staging

  const int tid = threadIdx.x, lane = tid & 31, w = tid >> 5;
  const int hl = lane >> 4, l15 = lane & 15;
  const int bx = blockIdx.x;
  const int qblk = bx % QB_;
  const int n = (bx / QB_) % NH_;
  const int b = bx / (QB_ * NH_);
  const int h = n / NREP_;
  const int q0 = qblk * 64 + w * 16;
  const int cur = *curp;
  const int stb = startind[b];
  const float scale = 0.08838834764831845f;  // 128^-0.5

  // persistent Q fragments (16 rows x HD=128 -> 4 A-frags)
  v16h qf[4];
  {
    const _Float16* qb = Qh + ((size_t)((b * T_ + q0 + l15) * NH_ + n)) * HD_;
    for (int d = 0; d < 4; ++d)
      qf[d] = hcat(*(const half8*)(qb + d * 32 + hl * 8),
                   *(const half8*)(qb + d * 32 + 16 + hl * 8));
  }
  int tq[8], sq[8];
  for (int r = 0; r < 8; ++r) { tq[r] = q0 + r + hl * 8; sq[r] = seg[b * T_ + tq[r]]; }

  float mrow[8], lrow[8];
  v8f of[8];
  for (int r = 0; r < 8; ++r) { mrow[r] = -__builtin_inff(); lrow[r] = 0.0f; }
  for (int c = 0; c < 8; ++c)
    for (int r = 0; r < 8; ++r) of[c][r] = 0.0f;

  _Float16* Pw = &Pt[w * 16 * PLD];
  int span = qblk * 64 + 64; if (span > T_) span = T_;
  const int s_hi = cur + span;                 // exclusive; only [0, cur+T) can pass the mask

  // loader geometry (128 threads x 32 halves each row-slice)
  const int ldr = tid >> 2;
  const int ldc = (tid & 3) * 32;
  const unsigned ksoff = lds_off32(&Ks[ldr * KLD + ldc]);

  for (int s0 = 0; s0 < s_hi; s0 += 32) {
    __syncthreads();   // prior iteration's tile reads complete before overwrite
    {
      int s = s0 + ldr;
      if (s < S_) {
        // K tile: straight copy -> async DMA into LDS (masked lanes skip; those
        // columns are unconditionally replaced by K_MASK later)
        const _Float16* kp = Kh + ((size_t)((b * S_ + s) * KVH_ + h)) * HD_ + ldc;
        for (int i = 0; i < 4; ++i) async_g2l_b128(ksoff + i * 16, kp + i * 8);
        // V tile: transposed store (element-wise), sync path
        const _Float16* vp = Vh + ((size_t)((b * S_ + s) * KVH_ + h)) * HD_ + ldc;
        half8 vv[4];
        for (int i = 0; i < 4; ++i) vv[i] = *(const half8*)(vp + i * 8);
        for (int i = 0; i < 4; ++i)
          for (int e = 0; e < 8; ++e) Vt[(ldc + i * 8 + e) * VLD + ldr] = vv[i][e];
      } else {
        for (int i = 0; i < 32; ++i) Vt[(ldc + i) * VLD + ldr] = (_Float16)0.0f;
      }
    }
    wait_async0();
    __syncthreads();

    // logits: 16 q-rows x 32 s-cols = two 16x16 C frags, 8 WMMAs over HD
    v8f sf0, sf1;
    for (int r = 0; r < 8; ++r) { sf0[r] = 0.0f; sf1[r] = 0.0f; }
    for (int d = 0; d < 4; ++d) {
      const _Float16* p0 = &Ks[l15 * KLD + d * 32 + hl * 8];
      const _Float16* p1 = &Ks[(16 + l15) * KLD + d * 32 + hl * 8];
      v16h kb0 = hcat(*(const half8*)p0, *(const half8*)(p0 + 16));
      v16h kb1 = hcat(*(const half8*)p1, *(const half8*)(p1 + 16));
      sf0 = wmma16(qf[d], kb0, sf0);
      sf1 = wmma16(qf[d], kb1, sf1);
    }

    // mask + online softmax (row stats via 16-lane xor shuffles)
    float alpha[8];
    for (int r = 0; r < 8; ++r) {
      int t = tq[r];
      float x0, x1;
      {
        int s = s0 + l15;
        int inseg = (s >= stb && s < cur + T_) ? 1 : 0;
        bool ok = (s <= cur + t) && (inseg == sq[r]);
        x0 = ok ? sf0[r] * scale : KMASK_;
      }
      {
        int s = s0 + 16 + l15;
        int inseg = (s >= stb && s < cur + T_) ? 1 : 0;
        bool ok = (s <= cur + t) && (inseg == sq[r]);
        x1 = ok ? sf1[r] * scale : KMASK_;
      }
      float mx = fmaxf(x0, x1);
      for (int o = 1; o < 16; o <<= 1) mx = fmaxf(mx, __shfl_xor(mx, o, 32));
      float mn = fmaxf(mrow[r], mx);
      float a  = __expf(mrow[r] - mn);
      float p0 = __expf(x0 - mn);
      float p1 = __expf(x1 - mn);
      float sum = p0 + p1;
      for (int o = 1; o < 16; o <<= 1) sum += __shfl_xor(sum, o, 32);
      lrow[r] = lrow[r] * a + sum;
      mrow[r] = mn;
      alpha[r] = a;
      Pw[(r + hl * 8) * PLD + l15]      = (_Float16)p0;  // C layout -> LDS
      Pw[(r + hl * 8) * PLD + 16 + l15] = (_Float16)p1;
    }
    for (int c = 0; c < 8; ++c)
      for (int r = 0; r < 8; ++r) of[c][r] *= alpha[r];

    // reload P as A-frag (wave-private region: in-order LDS, no block barrier needed)
    const _Float16* pp = &Pw[l15 * PLD];
    v16h pf = hcat(*(const half8*)(pp + hl * 8), *(const half8*)(pp + 16 + hl * 8));
    for (int c = 0; c < 8; ++c) {
      const _Float16* vb = &Vt[(c * 16 + l15) * VLD + hl * 8];
      v16h vf = hcat(*(const half8*)vb, *(const half8*)(vb + 16));
      of[c] = wmma16(pf, vf, of[c]);
    }
  }

  for (int c = 0; c < 8; ++c)
    for (int r = 0; r < 8; ++r) {
      float val = of[c][r] / lrow[r];
      Oh[((size_t)((b * T_ + tq[r]) * NH_ + n)) * HD_ + c * 16 + l15] = (_Float16)val;
    }
}

// ---------------- host launcher ----------------
extern "C" void kernel_launch(void* const* d_in, const int* in_sizes, int n_in,
                              void* d_out, int out_size, void* d_ws, size_t ws_size,
                              hipStream_t stream) {
  const float* x   = (const float*)d_in[0];
  const int*   seg = (const int*)d_in[1];
  const float* kc  = (const float*)d_in[2];
  const float* vc  = (const float*)d_in[3];
  const int*   cur = (const int*)d_in[4];
  const float* wq  = (const float*)d_in[5];
  const float* bq  = (const float*)d_in[6];
  const float* wk  = (const float*)d_in[7];
  const float* bk  = (const float*)d_in[8];
  const float* wv  = (const float*)d_in[9];
  const float* bv  = (const float*)d_in[10];
  const float* wo  = (const float*)d_in[11];
  float* out = (float*)d_out;

  char* ws = (char*)d_ws;
  size_t off = 0;
  auto alloc = [&](size_t bytes) -> void* {
    void* p = ws + off;
    off = (off + bytes + 255) & ~(size_t)255;
    return p;
  };
  _Float16* xh   = (_Float16*)alloc((size_t)M_ * EMB_ * 2);
  _Float16* wqt  = (_Float16*)alloc((size_t)EMB_ * QKD_ * 2);
  _Float16* wkt  = (_Float16*)alloc((size_t)EMB_ * KVD_ * 2);
  _Float16* wvt  = (_Float16*)alloc((size_t)EMB_ * KVD_ * 2);
  _Float16* wot  = (_Float16*)alloc((size_t)QKD_ * EMB_ * 2);
  float*    qraw = (float*)   alloc((size_t)M_ * QKD_ * 4);
  float*    kraw = (float*)   alloc((size_t)M_ * KVD_ * 4);
  float*    vraw = (float*)   alloc((size_t)M_ * KVD_ * 4);
  _Float16* Qh   = (_Float16*)alloc((size_t)M_ * QKD_ * 2);
  _Float16* Kh   = (_Float16*)alloc((size_t)B_ * S_ * KVD_ * 2);
  _Float16* Vh   = (_Float16*)alloc((size_t)B_ * S_ * KVD_ * 2);
  int*      pos  = (int*)     alloc((size_t)M_ * 4);
  int*      stn  = (int*)     alloc((size_t)B_ * 4);
  _Float16* Oh   = (_Float16*)qraw;  // qraw is dead after rope_write; reuse for attention out

  // 1) dtype conversion / weight transposition
  {
    size_t n = (size_t)M_ * EMB_;
    cvt_f16<<<(unsigned)((n + 255) / 256), 256, 0, stream>>>(x, xh, n);
  }
  cvt_transpose<<<(unsigned)(((size_t)EMB_ * QKD_ + 255) / 256), 256, 0, stream>>>(wq, wqt, EMB_, QKD_);
  cvt_transpose<<<(unsigned)(((size_t)EMB_ * KVD_ + 255) / 256), 256, 0, stream>>>(wk, wkt, EMB_, KVD_);
  cvt_transpose<<<(unsigned)(((size_t)EMB_ * KVD_ + 255) / 256), 256, 0, stream>>>(wv, wvt, EMB_, KVD_);
  cvt_transpose<<<(unsigned)(((size_t)QKD_ * EMB_ + 255) / 256), 256, 0, stream>>>(wo, wot, QKD_, EMB_);

  // 2) positions / left pads
  compute_pos<<<B_, 32, 0, stream>>>(seg, cur, pos, stn);

  // 3) QKV projections (WMMA GEMMs, bias fused)
  gemm_f16f32<<<dim3(QKD_ / 128, M_ / 128), 256, 0, stream>>>(xh, wqt, bq, qraw, M_, QKD_, EMB_);
  gemm_f16f32<<<dim3(KVD_ / 128, M_ / 128), 256, 0, stream>>>(xh, wkt, bk, kraw, M_, KVD_, EMB_);
  gemm_f16f32<<<dim3(KVD_ / 128, M_ / 128), 256, 0, stream>>>(xh, wvt, bv, vraw, M_, KVD_, EMB_);

  // 4) KV workspace: import old cache rows, then RoPE + scatter new rows
  {
    size_t n = (size_t)B_ * S_ * KVH_ * HD_;
    fill_cache<<<(unsigned)((n + 255) / 256), 256, 0, stream>>>(kc, vc, cur, Kh, Vh);
  }
  rope_write<<<B_ * T_, 256, 0, stream>>>(qraw, kraw, vraw, pos, cur, Qh, Kh, Vh);

  // 5) flash attention
  flash_attn<<<B_ * NH_ * QB_, 128, 0, stream>>>(Qh, Kh, Vh, seg, stn, cur, Oh);

  // 6) output projection -> d_out (f32)
  gemm_f16f32<<<dim3(EMB_ / 128, M_ / 128), 256, 0, stream>>>(Oh, wot, nullptr, out, M_, EMB_, QKD_);
}
